// STPGSR_25537875542503
// MI455X (gfx1250) — matrix operand order repaired and checked by reading, not compiled, not run on hardware
//
#include <hip/hip_runtime.h>
#include <hip/hip_bf16.h>
#include <math.h>

// ---------------------------------------------------------------------------
// STPGSR forward for gfx1250 (MI455X), wave32, WMMA-based GEMMs.
//
// d_in index map (setup_inputs() insertion order, params flattened in
// construction order):
//  0 x_src[160*160] 1 src_edge_index(int, unused: graph is K160 minus diag)
//  2 src_edge_attr[25440] 3 target_mat[268*268] 4 dual_edge_index(unused)
//  5..13  conv0: Wq,bq,Wk,bk,Wv,bv,Ws,bs,We
//  14..16 gn0: w,b,ms
//  17..25 conv1: Wq,bq,Wk,bk,Wv,bv,Ws,bs,We
//  26..28 gn1: w,b,ms
//  29 resW 30 resb   31 gcnW 32 gcnb
//  33..40 dconv: Wq,bq,Wk,bk,Wv,bv,Ws,bs (scalars)
//  41..43 dgn: w,b,ms   44 outW 45 outb 46 alpha
// ---------------------------------------------------------------------------

#define N_SRC 160
#define N_TGT 268
#define N_DUAL 35778   // 268*267/2

typedef __attribute__((ext_vector_type(16))) _Float16 v16h;
typedef __attribute__((ext_vector_type(8)))  float    v8f;
typedef __attribute__((ext_vector_type(8)))  unsigned v8u;

__device__ __forceinline__ unsigned pack_f16x2(float a, float b) {
  union { _Float16 h[2]; unsigned u; } t;
  t.h[0] = (_Float16)a; t.h[1] = (_Float16)b;
  return t.u;
}

// ---------------------------------------------------------------------------
// WMMA GEMM: C[M,N] = op(A) @ B (+ bias[n]), fp32 in/out, f16 compute.
// One wave per 32x32 tile: 2x2 accumulators, 4 v_wmma per 32-wide K step
// with A/B fragment reuse.  Interior tiles stage with zero bounds checks
// (wave-uniform test); edge tiles use clamped addresses + 0/1-mask multiply
// so loads stay unconditional (no exec-mask branching).
// __launch_bounds__(32,1): single wave per block, full VGPR budget -> no spills.
// TRANSA=0: A[M,K] row-major.  TRANSA=1: element(m,k) = A[k*M + m].
// ---------------------------------------------------------------------------
template <int TRANSA>
__global__ __launch_bounds__(32, 1)
void wmma_gemm_kernel(const float* __restrict__ A, const float* __restrict__ B,
                      const float* __restrict__ bias, float* __restrict__ C,
                      int M, int N, int K) {
  __shared__ unsigned Au[32][17];   // [m-in-tile][k-pair]   (k, k+1) packed f16x2
  __shared__ unsigned Bu[32][17];   // [n-in-tile][k-pair]   K-major so pairs contiguous
  const int lane = threadIdx.x;                 // 0..31 (wave32)
  const int tm = blockIdx.y * 32, tn = blockIdx.x * 32;
  const int lh = lane >> 4, lr = lane & 15;
  const int kp = lane & 15;                     // this lane's k-pair slot (loop-invariant)
  const int rb = lane >> 4;                     // row base; r = rb + 2t
  const bool inter_mn = (tm + 32 <= M) && (tn + 32 <= N);
  v8f acc[2][2] = {};
  for (int k0 = 0; k0 < K; k0 += 32) {
    const int gk = k0 + 2 * kp;
    if (inter_mn && (k0 + 32 <= K)) {
      // ---- fast path: no bounds checks at all
#pragma unroll 4
      for (int t = 0; t < 16; ++t) {
        const int r = rb + 2 * t;
        float a0, a1;
        if (TRANSA) {
          a0 = A[gk * M + (tm + r)];
          a1 = A[(gk + 1) * M + (tm + r)];
        } else {
          const float2 f = *reinterpret_cast<const float2*>(A + (size_t)(tm + r) * K + gk);
          a0 = f.x; a1 = f.y;
        }
        Au[r][kp] = pack_f16x2(a0, a1);
        Bu[r][kp] = pack_f16x2(B[gk * N + (tn + r)], B[(gk + 1) * N + (tn + r)]);
      }
    } else {
      // ---- edge path: clamped addresses, 0/1 mask multiply (branch-free)
      const int c0 = gk < K ? gk : K - 1;
      const int c1 = gk + 1 < K ? gk + 1 : K - 1;
      const float km0 = gk < K ? 1.f : 0.f;
      const float km1 = gk + 1 < K ? 1.f : 0.f;
#pragma unroll 4
      for (int t = 0; t < 16; ++t) {
        const int r = rb + 2 * t;
        const int gm = tm + r, gn = tn + r;
        const int cm = gm < M ? gm : M - 1;
        const int cn = gn < N ? gn : N - 1;
        const float mm = gm < M ? 1.f : 0.f;
        const float nm = gn < N ? 1.f : 0.f;
        const float a0 = (TRANSA ? A[c0 * M + cm] : A[cm * K + c0]) * (mm * km0);
        const float a1 = (TRANSA ? A[c1 * M + cm] : A[cm * K + c1]) * (mm * km1);
        Au[r][kp] = pack_f16x2(a0, a1);
        Bu[r][kp] = pack_f16x2(B[c0 * N + cn] * (nm * km0), B[c1 * N + cn] * (nm * km1));
      }
    }
    __syncthreads();
    // Fragment packing per CDNA5 ISA (CDNA5 ISA):
    //  A 16x32 f16: lane M = lane&15; VGPR v -> K = ((v<4)?0:16) + lh*8 + 2(v&3)
    //  B 32x16 f16: lane N = lane&15; VGPR v -> K = ((v<4)?0:16) + lh*8 + 2(v&3)
    v16h af[2], bf[2];
#pragma unroll
    for (int i = 0; i < 2; ++i) {
      v8u ua, ub;
#pragma unroll
      for (int v = 0; v < 8; ++v) {
        const int pa = ((v < 4) ? 0 : 8) + lh * 4 + (v & 3);   // A k-pair index
        ua[v] = Au[16 * i + lr][pa];
        ub[v] = Bu[16 * i + lr][pa];                   // B k-pair index
      }
      af[i] = __builtin_bit_cast(v16h, ua);
      bf[i] = __builtin_bit_cast(v16h, ub);
    }
    acc[0][0] = __builtin_amdgcn_wmma_f32_16x16x32_f16(false, af[0], false, bf[0], (short)0, acc[0][0], false, false);
    acc[0][1] = __builtin_amdgcn_wmma_f32_16x16x32_f16(false, af[0], false, bf[1], (short)0, acc[0][1], false, false);
    acc[1][0] = __builtin_amdgcn_wmma_f32_16x16x32_f16(false, af[1], false, bf[0], (short)0, acc[1][0], false, false);
    acc[1][1] = __builtin_amdgcn_wmma_f32_16x16x32_f16(false, af[1], false, bf[1], (short)0, acc[1][1], false, false);
    __syncthreads();
  }
  // C/D layout: VGPR r -> M = r + (lane>=16 ? 8 : 0), N = lane&15
#pragma unroll
  for (int i = 0; i < 2; ++i)
#pragma unroll
    for (int j = 0; j < 2; ++j)
#pragma unroll
      for (int r = 0; r < 8; ++r) {
        const int gm = tm + 16 * i + r + lh * 8, gn = tn + 16 * j + lr;
        if (gm < M && gn < N) {
          float v = acc[i][j][r];
          if (bias) v += bias[gn];
          C[gm * N + gn] = v;
        }
      }
}

// ---------------------------------------------------------------------------
// Dense TransformerConv attention on the fully-connected source graph
// (diagonal masked via alpha[d]=0).
// edge id for (src=s, dst=d): e = s*(n-1) + (d>s ? d-1 : d).
// ---------------------------------------------------------------------------
__global__ void tconv_attn_kernel(const float* __restrict__ q, const float* __restrict__ k,
                                  const float* __restrict__ v, const float* __restrict__ attr,
                                  const float* __restrict__ We, float* __restrict__ out,
                                  int n, int H, int C) {
  const int d = blockIdx.x, h = blockIdx.y, HC = H * C, tid = threadIdx.x;
  __shared__ float qd[96], lg[192], at[192], red[256];
  for (int c = tid; c < C; c += blockDim.x) qd[c] = q[d * HC + h * C + c];
  __syncthreads();
  float part = 0.f;
  for (int c = tid; c < C; c += blockDim.x) part += qd[c] * We[h * C + c];
  red[tid] = part; __syncthreads();
  for (int st = 128; st > 0; st >>= 1) { if (tid < st) red[tid] += red[tid + st]; __syncthreads(); }
  const float wq = red[0]; __syncthreads();
  const float rsC = rsqrtf((float)C);
  const int s = tid;
  if (s < n) {
    if (s == d) { lg[s] = -1e30f; at[s] = 0.f; }
    else {
      float a_sd = attr[s * (n - 1) + (d > s ? d - 1 : d)];
      float dot = 0.f;
      for (int c = 0; c < C; ++c) dot += qd[c] * k[s * HC + h * C + c];
      lg[s] = (dot + a_sd * wq) * rsC;
      at[s] = a_sd;
    }
  }
  __syncthreads();
  red[tid] = (s < n) ? lg[s] : -1e30f; __syncthreads();
  for (int st = 128; st > 0; st >>= 1) { if (tid < st) red[tid] = fmaxf(red[tid], red[tid + st]); __syncthreads(); }
  const float mx = red[0]; __syncthreads();
  float ex = 0.f;
  if (s < n && s != d) ex = __expf(lg[s] - mx);
  red[tid] = ex; __syncthreads();
  for (int st = 128; st > 0; st >>= 1) { if (tid < st) red[tid] += red[tid + st]; __syncthreads(); }
  const float ssum = red[0] + 1e-16f; __syncthreads();
  if (s < n) lg[s] = ex / ssum;                 // alpha; alpha[d] == 0
  __syncthreads();
  for (int c = tid; c < C; c += blockDim.x) {
    float acc = 0.f, wec = We[h * C + c];
    for (int ss = 0; ss < n; ++ss)              // no skip: alpha[d]=0 kills diag term
      acc += lg[ss] * (v[ss * HC + h * C + c] + at[ss] * wec);
    out[d * HC + h * C + c] += acc;
  }
}

// GraphNorm (over rows) per column + ReLU, optional elementwise skip input.
__global__ void gnorm_relu_cols(const float* __restrict__ x, const float* __restrict__ skip,
                                const float* __restrict__ w, const float* __restrict__ b,
                                const float* __restrict__ ms, float* __restrict__ y,
                                int n, int d) {
  const int col = blockIdx.x, tid = threadIdx.x;
  __shared__ float red[256];
  float v = 0.f;
  if (tid < n) { v = x[tid * d + col]; if (skip) v += skip[tid * d + col]; }
  red[tid] = v; __syncthreads();
  for (int st = 128; st > 0; st >>= 1) { if (tid < st) red[tid] += red[tid + st]; __syncthreads(); }
  const float mean = red[0] / (float)n; __syncthreads();
  const float o = v - mean * ms[col];
  red[tid] = (tid < n) ? o * o : 0.f; __syncthreads();
  for (int st = 128; st > 0; st >>= 1) { if (tid < st) red[tid] += red[tid + st]; __syncthreads(); }
  const float var = red[0] / (float)n; __syncthreads();
  if (tid < n) y[tid * d + col] = fmaxf(0.f, w[col] * o * rsqrtf(var + 1e-5f) + b[col]);
}

// GCN degree^-1/2 (improved=True self-loop weight 2.0).
__global__ void gcn_dinv_kernel(const float* __restrict__ attr, float* __restrict__ dinv, int n) {
  const int d = blockIdx.x * blockDim.x + threadIdx.x;
  if (d >= n) return;
  float s = 2.0f;
  for (int src = 0; src < n; ++src) {
    const float msk = (src == d) ? 0.f : 1.f;
    s += attr[src * (n - 1) + (d > src ? d - 1 : d)] * msk;
  }
  dinv[d] = rsqrtf(s);
}

// Dense normalized adjacency A[d,s] = dinv[d]*w(s->d)*dinv[s], diag 0.
__global__ void gcn_amat_kernel(const float* __restrict__ attr, const float* __restrict__ dinv,
                                float* __restrict__ Am, int n) {
  const int d = blockIdx.x, s = threadIdx.x;
  if (s < n)
    Am[d * n + s] = (s == d) ? 0.f
                             : dinv[d] * attr[s * (n - 1) + (d > s ? d - 1 : d)] * dinv[s];
}

__global__ void gcn_combine_kernel(const float* __restrict__ agg, const float* __restrict__ xl,
                                   const float* __restrict__ dinv, const float* __restrict__ b,
                                   float* __restrict__ y, int n, int dcols) {
  const int idx = blockIdx.x * blockDim.x + threadIdx.x;
  if (idx >= n * dcols) return;
  const int i = idx / dcols, c = idx - i * dcols;
  const float di = dinv[i];
  y[idx] = agg[idx] + 2.f * di * di * xl[idx] + b[c];
}

__global__ void reduce_minmax_kernel(const float* __restrict__ x, float* __restrict__ omn,
                                     float* __restrict__ omx, int n) {
  __shared__ float smn[512], smx[512];
  const int tid = threadIdx.x;
  float mn = 1e30f, mx = -1e30f;
  for (int i = tid; i < n; i += blockDim.x) { float v = x[i]; mn = fminf(mn, v); mx = fmaxf(mx, v); }
  smn[tid] = mn; smx[tid] = mx; __syncthreads();
  for (int st = 256; st > 0; st >>= 1) {
    if (tid < st) { smn[tid] = fminf(smn[tid], smn[tid + st]); smx[tid] = fmaxf(smx[tid], smx[tid + st]); }
    __syncthreads();
  }
  if (tid == 0) { *omn = smn[0]; *omx = smx[0]; }
}

// Normalize xt -> xtn (dense, symmetric) and scatter upper-tri into init_pred.
__global__ void extract_kernel(const float* __restrict__ xt, const float* __restrict__ mn,
                               const float* __restrict__ mx, float* __restrict__ xtn,
                               float* __restrict__ init, int n) {
  const int idx = blockIdx.x * blockDim.x + threadIdx.x;
  if (idx >= n * n) return;
  const float lo = *mn, rng = *mx - lo + 1e-8f;
  const float v = (xt[idx] - lo) / rng;
  xtn[idx] = v;
  const int i = idx / n, j = idx - i * n;
  if (i < j) init[i * n - (i * (i + 1)) / 2 + (j - i - 1)] = v;
}

// ---------------------------------------------------------------------------
// Dual-graph TransformerConv (H=1,C=1), dense reformulation; branch-free
// inner loops (select to -inf / multiply-by-mask).  Avoids reading the
// 19M-entry (152 MB) dual edge list entirely.
// ---------------------------------------------------------------------------
__global__ void dual_conv_kernel(const float* __restrict__ xtn,
                                 const float* __restrict__ pWq, const float* __restrict__ pbq,
                                 const float* __restrict__ pWk, const float* __restrict__ pbk,
                                 const float* __restrict__ pWv, const float* __restrict__ pbv,
                                 const float* __restrict__ pWs, const float* __restrict__ pbs,
                                 float* __restrict__ xdpre, int n) {
  const int i = blockIdx.x;
  __shared__ float rowi[N_TGT];
  for (int m = threadIdx.x; m < n; m += blockDim.x) rowi[m] = xtn[i * n + m];
  __syncthreads();
  const float wq = *pWq, bq = *pbq, wk = *pWk, bk = *pbk;
  const float wv = *pWv, bv = *pbv, wss = *pWs, bs = *pbs;
  for (int j = i + 1 + (int)threadIdx.x; j < n; j += blockDim.x) {
    __builtin_prefetch(&xtn[j * n], 0, 1);     // global_prefetch_b8
    const float p = rowi[j];
    const float t = wq * p + bq;
    float mxl = -1e30f;
    for (int m = 0; m < n; ++m) {
      const bool sk = (m == i) | (m == j);
      const float l1 = t * (wk * rowi[m] + bk);
      const float l2 = t * (wk * xtn[j * n + m] + bk);
      const float l = fmaxf(l1, l2);
      mxl = fmaxf(mxl, sk ? -1e30f : l);
    }
    float se = 0.f, sev = 0.f;
    for (int m = 0; m < n; ++m) {
      const float msk = ((m == i) | (m == j)) ? 0.f : 1.f;
      const float p1 = rowi[m], p2 = xtn[j * n + m];
      const float e1 = __expf(t * (wk * p1 + bk) - mxl) * msk;
      const float e2 = __expf(t * (wk * p2 + bk) - mxl) * msk;
      se += e1 + e2;
      sev += e1 * (wv * p1 + bv) + e2 * (wv * p2 + bv);
    }
    const int tt = i * n - (i * (i + 1)) / 2 + (j - i - 1);
    xdpre[tt] = sev / (se + 1e-16f) + wss * p + bs;
  }
}

__global__ void reduce_mean2_kernel(const float* __restrict__ a, const float* __restrict__ b,
                                    float* __restrict__ out, int n) {
  __shared__ float red[512];
  const int tid = threadIdx.x;
  float s = 0.f;
  for (int i = tid; i < n; i += blockDim.x) s += a[i] + b[i];
  red[tid] = s; __syncthreads();
  for (int st = 256; st > 0; st >>= 1) { if (tid < st) red[tid] += red[tid + st]; __syncthreads(); }
  if (tid == 0) *out = red[0] / (float)n;
}

__global__ void reduce_var2_kernel(const float* __restrict__ a, const float* __restrict__ b,
                                   const float* __restrict__ mean, const float* __restrict__ ms,
                                   float* __restrict__ out, int n) {
  __shared__ float red[512];
  const int tid = threadIdx.x;
  const float msh = (*mean) * (*ms);
  float s = 0.f;
  for (int i = tid; i < n; i += blockDim.x) { const float o = a[i] + b[i] - msh; s += o * o; }
  red[tid] = s; __syncthreads();
  for (int st = 256; st > 0; st >>= 1) { if (tid < st) red[tid] += red[tid + st]; __syncthreads(); }
  if (tid == 0) *out = red[0] / (float)n;
}

// y = relu(GraphNorm(xdpre+init)); u = y*outW + outb
__global__ void dual_post_kernel(const float* __restrict__ xdpre, const float* __restrict__ init,
                                 const float* __restrict__ mean, const float* __restrict__ var,
                                 const float* __restrict__ w, const float* __restrict__ b,
                                 const float* __restrict__ ms, const float* __restrict__ outW,
                                 const float* __restrict__ outb, float* __restrict__ u, int n) {
  const int i = blockIdx.x * blockDim.x + threadIdx.x;
  if (i >= n) return;
  const float o = xdpre[i] + init[i] - (*mean) * (*ms);
  const float y = fmaxf(0.f, (*w) * o * rsqrtf(*var + 1e-5f) + (*b));
  u[i] = y * (*outW) + (*outb);
}

__global__ void final_out_kernel(const float* __restrict__ u, const float* __restrict__ init,
                                 const float* __restrict__ target, const float* __restrict__ mn,
                                 const float* __restrict__ mx, const float* __restrict__ alpha,
                                 float* __restrict__ out, int n, int nd) {
  const int i = blockIdx.x;
  const float a = 1.f / (1.f + __expf(-*alpha));
  const float lo = *mn, rng = *mx - lo + 1e-8f;
  for (int j = i + 1 + (int)threadIdx.x; j < n; j += blockDim.x) {
    const int t = i * n - (i * (i + 1)) / 2 + (j - i - 1);
    const float xd = (u[t] - lo) / rng;
    out[t] = a * xd + (1.f - a) * init[t];
    out[nd + t] = target[i * n + j];
  }
}

// ---------------------------------------------------------------------------
static inline void launch_gemm(const float* A, const float* B, const float* bias, float* C,
                               int M, int N, int K, int transA, hipStream_t s) {
  dim3 g((N + 31) / 32, (M + 31) / 32);
  if (transA) wmma_gemm_kernel<1><<<g, 32, 0, s>>>(A, B, bias, C, M, N, K);
  else        wmma_gemm_kernel<0><<<g, 32, 0, s>>>(A, B, bias, C, M, N, K);
}

extern "C" void kernel_launch(void* const* d_in, const int* in_sizes, int n_in,
                              void* d_out, int out_size, void* d_ws, size_t ws_size,
                              hipStream_t stream) {
  const float* x_src  = (const float*)d_in[0];
  const float* attr   = (const float*)d_in[2];
  const float* target = (const float*)d_in[3];
  // conv0
  const float *Wq0 = (const float*)d_in[5],  *bq0 = (const float*)d_in[6];
  const float *Wk0 = (const float*)d_in[7],  *bk0 = (const float*)d_in[8];
  const float *Wv0 = (const float*)d_in[9],  *bv0 = (const float*)d_in[10];
  const float *Ws0 = (const float*)d_in[11], *bs0 = (const float*)d_in[12];
  const float *We0 = (const float*)d_in[13];
  const float *gn0w = (const float*)d_in[14], *gn0b = (const float*)d_in[15], *gn0m = (const float*)d_in[16];
  // conv1
  const float *Wq1 = (const float*)d_in[17], *bq1 = (const float*)d_in[18];
  const float *Wk1 = (const float*)d_in[19], *bk1 = (const float*)d_in[20];
  const float *Wv1 = (const float*)d_in[21], *bv1 = (const float*)d_in[22];
  const float *Ws1 = (const float*)d_in[23], *bs1 = (const float*)d_in[24];
  const float *We1 = (const float*)d_in[25];
  const float *gn1w = (const float*)d_in[26], *gn1b = (const float*)d_in[27], *gn1m = (const float*)d_in[28];
  const float *resW = (const float*)d_in[29], *resb = (const float*)d_in[30];
  const float *gcnW = (const float*)d_in[31], *gcnb = (const float*)d_in[32];
  const float *dWq = (const float*)d_in[33], *dbq = (const float*)d_in[34];
  const float *dWk = (const float*)d_in[35], *dbk = (const float*)d_in[36];
  const float *dWv = (const float*)d_in[37], *dbv = (const float*)d_in[38];
  const float *dWs = (const float*)d_in[39], *dbs = (const float*)d_in[40];
  const float *dgnw = (const float*)d_in[41], *dgnb = (const float*)d_in[42], *dgnm = (const float*)d_in[43];
  const float *outW = (const float*)d_in[44], *outb = (const float*)d_in[45];
  const float *alpha = (const float*)d_in[46];
  float* out = (float*)d_out;

  // Workspace carve-up (floats); total ~2.7 MB (fits easily in L2).
  float* ws = (float*)d_ws;
  size_t off = 0;
  auto take = [&](size_t nf) { float* p = ws + off; off += nf; return p; };
  float* X_RES = take(160 * 268);
  float* Q0 = take(160 * 64); float* K0 = take(160 * 64); float* V0 = take(160 * 64);
  float* OUT0 = take(160 * 64); float* H1 = take(160 * 64);
  float* Q1 = take(160 * 268); float* K1 = take(160 * 268); float* V1 = take(160 * 268);
  float* OUT1 = take(160 * 268); float* X2 = take(160 * 268);
  float* XL = take(160 * 268); float* AMAT = take(160 * 160);
  float* AGG = take(160 * 268); float* Y = take(160 * 268);
  float* XT = take(268 * 268); float* XTN = take(268 * 268);
  float* DINV = take(160);
  float* INIT = take(N_DUAL); float* XDPRE = take(N_DUAL); float* U = take(N_DUAL);
  float* SCAL = take(8);   // [0]=xt min [1]=xt max [2]=mean [3]=var [4]=u min [5]=u max

  // --- residual + conv0 projections (WMMA GEMMs) ---
  launch_gemm(x_src, resW, resb, X_RES, 160, 268, 160, 0, stream);
  launch_gemm(x_src, Wq0, bq0, Q0, 160, 64, 160, 0, stream);
  launch_gemm(x_src, Wk0, bk0, K0, 160, 64, 160, 0, stream);
  launch_gemm(x_src, Wv0, bv0, V0, 160, 64, 160, 0, stream);
  launch_gemm(x_src, Ws0, bs0, OUT0, 160, 64, 160, 0, stream);   // skip term
  tconv_attn_kernel<<<dim3(160, 4), 256, 0, stream>>>(Q0, K0, V0, attr, We0, OUT0, 160, 4, 16);
  gnorm_relu_cols<<<64, 256, 0, stream>>>(OUT0, nullptr, gn0w, gn0b, gn0m, H1, 160, 64);

  // --- conv1 ---
  launch_gemm(H1, Wq1, bq1, Q1, 160, 268, 64, 0, stream);
  launch_gemm(H1, Wk1, bk1, K1, 160, 268, 64, 0, stream);
  launch_gemm(H1, Wv1, bv1, V1, 160, 268, 64, 0, stream);
  launch_gemm(H1, Ws1, bs1, OUT1, 160, 268, 64, 0, stream);
  tconv_attn_kernel<<<dim3(160, 4), 256, 0, stream>>>(Q1, K1, V1, attr, We1, OUT1, 160, 4, 67);
  gnorm_relu_cols<<<268, 256, 0, stream>>>(OUT1, X_RES, gn1w, gn1b, gn1m, X2, 160, 268);

  // --- GCN (dense adjacency, WMMA for both GEMMs) ---
  launch_gemm(X2, gcnW, nullptr, XL, 160, 268, 268, 0, stream);
  gcn_dinv_kernel<<<1, 192, 0, stream>>>(attr, DINV, 160);
  gcn_amat_kernel<<<160, 192, 0, stream>>>(attr, DINV, AMAT, 160);
  launch_gemm(AMAT, XL, nullptr, AGG, 160, 268, 160, 0, stream);
  gcn_combine_kernel<<<(160 * 268 + 255) / 256, 256, 0, stream>>>(AGG, XL, DINV, gcnb, Y, 160, 268);

  // --- xt = Y^T @ Y (transA WMMA), min-max, upper-tri extraction ---
  launch_gemm(Y, Y, nullptr, XT, 268, 268, 160, 1, stream);
  reduce_minmax_kernel<<<1, 512, 0, stream>>>(XT, SCAL + 0, SCAL + 1, 268 * 268);
  extract_kernel<<<(268 * 268 + 255) / 256, 256, 0, stream>>>(XT, SCAL + 0, SCAL + 1, XTN, INIT, 268);

  // --- dual-graph conv (dense, no 152 MB edge list needed) ---
  dual_conv_kernel<<<268, 256, 0, stream>>>(XTN, dWq, dbq, dWk, dbk, dWv, dbv, dWs, dbs, XDPRE, 268);

  // --- dual GraphNorm + relu + linear + minmax + blend ---
  reduce_mean2_kernel<<<1, 512, 0, stream>>>(XDPRE, INIT, SCAL + 2, N_DUAL);
  reduce_var2_kernel<<<1, 512, 0, stream>>>(XDPRE, INIT, SCAL + 2, dgnm, SCAL + 3, N_DUAL);
  dual_post_kernel<<<(N_DUAL + 255) / 256, 256, 0, stream>>>(XDPRE, INIT, SCAL + 2, SCAL + 3,
                                                             dgnw, dgnb, dgnm, outW, outb, U, N_DUAL);
  reduce_minmax_kernel<<<1, 512, 0, stream>>>(U, SCAL + 4, SCAL + 5, N_DUAL);
  final_out_kernel<<<268, 256, 0, stream>>>(U, INIT, target, SCAL + 4, SCAL + 5, alpha,
                                            out, 268, N_DUAL);
  (void)in_sizes; (void)n_in; (void)out_size; (void)ws_size;
}
